// JSONTreeLSTM_33904471834913
// MI455X (gfx1250) — matrix operand order repaired and checked by reading, not compiled, not run on hardware
//
#include <hip/hip_runtime.h>
#include <math.h>

#define MEMD 128
#define KOBJ 4096
#define LARR 128

typedef __attribute__((ext_vector_type(16))) _Float16 v16h;
typedef __attribute__((ext_vector_type(8)))  _Float16 h8;
typedef __attribute__((ext_vector_type(8)))  float    v8f;

union V16 { v16h v; h8 h[2]; };

#define LOG2E 1.4426950408889634f

// Branch-free activations on v_exp_f32 / v_rcp_f32 (no EXEC divergence,
// no IEEE div sequence). ftanh's exp2 arg is always <= 0; sigm's can be positive.
__device__ __forceinline__ float sigm(float x) {
    return __builtin_amdgcn_rcpf(1.0f + __builtin_amdgcn_exp2f(-LOG2E * x));
}
__device__ __forceinline__ float ftanh(float x) {
    float ax = __builtin_fabsf(x);
    float e  = __builtin_amdgcn_exp2f(-2.0f * LOG2E * ax);
    float r  = (1.0f - e) * __builtin_amdgcn_rcpf(1.0f + e);
    return __builtin_copysignf(r, x);
}

// ---- workspace layout (byte offsets) ----
#define WS_MEAN   0                  // 100 f32
#define WS_STD    512                // 100 f32
#define WS_U      1024               // 512 f32  : W_ih[:,128:] @ w_num
#define WS_CC     3072               // 512 f32  : W_ih[:,128:] @ b_num + b_ih + b_hh
#define WS_FC     5120               // 128 f32  : sum_k sigm(f)*c
#define WS_HS     5632               // 128 f32  : sum_k h
#define WS_WH     6144               // 512*128 f16 (131072 B)
#define WS_WFH    (6144 + 131072)    // 128*128 f16 (32768 B)

// ---------------------------------------------------------------------------
// Kernel 1: precompute rank-1 input projection, f16 weight copies, number stats
// ---------------------------------------------------------------------------
__global__ void prep_kernel(const float* __restrict__ numbers,
                            const float* __restrict__ w_num,
                            const float* __restrict__ b_num,
                            const float* __restrict__ W_ih,
                            const float* __restrict__ W_hh,
                            const float* __restrict__ b_ih,
                            const float* __restrict__ b_hh,
                            const float* __restrict__ W_fh,
                            char* __restrict__ ws) {
    const int tid = threadIdx.x;  // 512 threads
    float*     meanN = (float*)(ws + WS_MEAN);
    float*     stdN  = (float*)(ws + WS_STD);
    float*     u     = (float*)(ws + WS_U);
    float*     cc    = (float*)(ws + WS_CC);
    float*     fc    = (float*)(ws + WS_FC);
    float*     hs    = (float*)(ws + WS_HS);
    _Float16*  Wh    = (_Float16*)(ws + WS_WH);
    _Float16*  Wf    = (_Float16*)(ws + WS_WFH);

    // u[n], cc[n] over the used (right) half of W_ih
    {
        const float* row = W_ih + tid * 2 * MEMD + MEMD;
        float su = 0.f, sc = 0.f;
        for (int m = 0; m < MEMD; ++m) { su += row[m] * w_num[m]; sc += row[m] * b_num[m]; }
        u[tid]  = su;
        cc[tid] = sc + b_ih[tid] + b_hh[tid];
    }
    // f16 copies of recurrence / gating weights
    for (int i = tid; i < 4 * MEMD * MEMD; i += 512) Wh[i] = (_Float16)W_hh[i];
    for (int i = tid; i < MEMD * MEMD; i += 512)     Wf[i] = (_Float16)W_fh[i];
    if (tid < MEMD) { fc[tid] = 0.f; hs[tid] = 0.f; }
    // running number stats: cap at 100, frozen afterwards (index 99)
    if (tid == 0) {
        float cs = 0.f, css = 0.f;
        for (int i = 0; i < 100; ++i) {
            float x = numbers[i];
            cs += x; css += x * x;
            float n = (float)(i + 1);
            float mean = cs / n;
            float var  = css / n - mean * mean;
            meanN[i] = mean;
            stdN[i]  = sqrtf(fmaxf(var, 0.f));
        }
    }
}

// ---------------------------------------------------------------------------
// Kernel 2: LSTM recurrence over 128 steps, 16 sequences per workgroup.
// 8 waves; wave w owns hidden columns [16w,16w+16). All WMMA f32_16x16x32_f16.
// ---------------------------------------------------------------------------
__global__ void __launch_bounds__(256)
lstm_kernel(const float* __restrict__ numbers,
            const float* __restrict__ b_fh,
            char* __restrict__ ws) {
    __shared__ __align__(16) _Float16 hbuf[2][16 * MEMD];  // double-buffered h (f16)
    __shared__ float xn[LARR * 16];                        // normalized inputs [t][row]

    const float*    meanN = (const float*)(ws + WS_MEAN);
    const float*    stdN  = (const float*)(ws + WS_STD);
    const float*    u     = (const float*)(ws + WS_U);
    const float*    ccst  = (const float*)(ws + WS_CC);
    float*          fc    = (float*)(ws + WS_FC);
    float*          hs    = (float*)(ws + WS_HS);
    const _Float16* Wh    = (const _Float16*)(ws + WS_WH);
    const _Float16* Wf    = (const _Float16*)(ws + WS_WFH);

    const int tid  = threadIdx.x;
    const int wave = tid >> 5;
    const int lane = tid & 31;
    const int c16  = lane & 15;   // column within 16-wide tile / row M for A-frags
    const int hi   = lane >> 4;   // lane half-group
    const int col  = wave * 16 + c16;        // hidden column this lane owns
    const int krow0 = blockIdx.x * 16;       // first sequence of this tile

    // --- normalized inputs for the whole tile ---
    for (int i = tid; i < 16 * LARR; i += 256) {
        int t = i >> 4, row = i & 15;
        int idx = (krow0 + row) * LARR + t;  // traversal order = flattened row-major
        float x = numbers[idx];
        int j = idx < 100 ? idx : 99;        // stats frozen after 100 numbers
        float mean = meanN[j], sd = stdN[j];
        float y = x - mean;
        bool use = (idx >= 3) && (sd > 1e-8f);
        xn[t * 16 + row] = use ? y / sd : y;
    }
    for (int i = tid; i < 16 * MEMD; i += 256) hbuf[0][i] = (_Float16)0.f;

    // --- preload B fragments of W_hh (held in registers for all 128 steps) ---
    // B(32x16): lane col N = c16, V0-3: K=8*hi.., V4-7: K=16+8*hi.. (pairs per VGPR)
    V16 B[4][4];
    float ug[4], cg[4];
#pragma unroll
    for (int g = 0; g < 4; ++g) {            // gate order: i, f, g, o
        int n = g * MEMD + col;
        ug[g] = u[n];
        cg[g] = ccst[n];
#pragma unroll
        for (int kk = 0; kk < 4; ++kk) {
            int base = n * MEMD + kk * 32 + hi * 8;
            B[g][kk].h[0] = *(const h8*)(Wh + base);
            B[g][kk].h[1] = *(const h8*)(Wh + base + 16);
        }
    }
    float creg[8];
#pragma unroll
    for (int v = 0; v < 8; ++v) creg[v] = 0.f;

    __syncthreads();

    // --- sequential scan: gates = h @ W_hh.T + xn*u + const ---
    for (int t = 0; t < LARR; ++t) {
        const _Float16* hcur = hbuf[t & 1];
        _Float16*       hnxt = hbuf[(t + 1) & 1];
        v8f acc0 = {}, acc1 = {}, acc2 = {}, acc3 = {};
#pragma unroll
        for (int kk = 0; kk < 4; ++kk) {
            // A(16x32): lane row M = c16; V0-3: K=8*hi.., V4-7: K=16+8*hi..
            V16 A;
            int a0 = c16 * MEMD + kk * 32 + hi * 8;
            A.h[0] = *(const h8*)(hcur + a0);
            A.h[1] = *(const h8*)(hcur + a0 + 16);
            acc0 = __builtin_amdgcn_wmma_f32_16x16x32_f16(false, A.v, false, B[0][kk].v, (short)0, acc0, false, false);
            acc1 = __builtin_amdgcn_wmma_f32_16x16x32_f16(false, A.v, false, B[1][kk].v, (short)0, acc1, false, false);
            acc2 = __builtin_amdgcn_wmma_f32_16x16x32_f16(false, A.v, false, B[2][kk].v, (short)0, acc2, false, false);
            acc3 = __builtin_amdgcn_wmma_f32_16x16x32_f16(false, A.v, false, B[3][kk].v, (short)0, acc3, false, false);
        }
#pragma unroll
        for (int v = 0; v < 8; ++v) {
            int row = v + hi * 8;            // C/D layout: VGPR v, lane group -> M
            float xv = xn[t * 16 + row];
            float gi = acc0[v] + xv * ug[0] + cg[0];
            float gf = acc1[v] + xv * ug[1] + cg[1];
            float gg = acc2[v] + xv * ug[2] + cg[2];
            float go = acc3[v] + xv * ug[3] + cg[3];
            float cn = sigm(gf) * creg[v] + sigm(gi) * ftanh(gg);
            creg[v] = cn;
            hnxt[row * MEMD + col] = (_Float16)(sigm(go) * ftanh(cn));
        }
        __syncthreads();
    }

    // --- child-sum gating: f_g = sigm(h @ W_fh.T + b_fh); reduce over all K ---
    const _Float16* hfin = hbuf[0];          // after 128 steps final h is buffer 0
    v8f accf = {};
#pragma unroll
    for (int kk = 0; kk < 4; ++kk) {
        V16 A, Bf;
        int a0 = c16 * MEMD + kk * 32 + hi * 8;
        A.h[0] = *(const h8*)(hfin + a0);
        A.h[1] = *(const h8*)(hfin + a0 + 16);
        int base = col * MEMD + kk * 32 + hi * 8;
        Bf.h[0] = *(const h8*)(Wf + base);
        Bf.h[1] = *(const h8*)(Wf + base + 16);
        accf = __builtin_amdgcn_wmma_f32_16x16x32_f16(false, A.v, false, Bf.v, (short)0, accf, false, false);
    }
    float bf = b_fh[col];
    float pfc = 0.f, phs = 0.f;
#pragma unroll
    for (int v = 0; v < 8; ++v) {
        int row = v + hi * 8;
        pfc += sigm(accf[v] + bf) * creg[v];
        phs += (float)hfin[row * MEMD + col];
    }
    atomicAdd(&fc[col], pfc);
    atomicAdd(&hs[col], phs);
}

// ---------------------------------------------------------------------------
// Kernel 3: root Tree-LSTM cell + output projection (tiny: 1x128 vectors)
// ---------------------------------------------------------------------------
__global__ void root_kernel(const float* __restrict__ W_iouh,
                            const float* __restrict__ b_iouh,
                            const float* __restrict__ W_lout,
                            const float* __restrict__ b_lout,
                            const char* __restrict__ ws,
                            float* __restrict__ out) {
    __shared__ float iou[3 * MEMD];
    __shared__ float hroot[MEMD];
    const float* fc  = (const float*)(ws + WS_FC);
    const float* hsb = (const float*)(ws + WS_HS);
    const int tid = threadIdx.x;  // 384 threads
    {
        const float* wrow = W_iouh + tid * MEMD;
        float s = b_iouh[tid];
        for (int m = 0; m < MEMD; ++m) s += hsb[m] * wrow[m];
        iou[tid] = s;
    }
    __syncthreads();
    if (tid < MEMD) {
        float gi = iou[tid], go = iou[MEMD + tid], gu = iou[2 * MEMD + tid];
        float c = sigm(gi) * ftanh(gu) + fc[tid];
        out[tid] = c;
        hroot[tid] = sigm(go) * ftanh(c);
    }
    __syncthreads();
    if (tid < MEMD) {
        const float* wrow = W_lout + tid * MEMD;
        float s = b_lout[tid];
        for (int m = 0; m < MEMD; ++m) s += hroot[m] * wrow[m];
        out[MEMD + tid] = s;
    }
}

// ---------------------------------------------------------------------------
extern "C" void kernel_launch(void* const* d_in, const int* in_sizes, int n_in,
                              void* d_out, int out_size, void* d_ws, size_t ws_size,
                              hipStream_t stream) {
    const float* numbers = (const float*)d_in[0];
    const float* w_num   = (const float*)d_in[1];
    const float* b_num   = (const float*)d_in[2];
    const float* W_ih    = (const float*)d_in[3];
    const float* W_hh    = (const float*)d_in[4];
    const float* b_ih    = (const float*)d_in[5];
    const float* b_hh    = (const float*)d_in[6];
    const float* W_fh    = (const float*)d_in[7];
    const float* b_fh    = (const float*)d_in[8];
    const float* W_iouh  = (const float*)d_in[9];
    const float* b_iouh  = (const float*)d_in[10];
    const float* W_lout  = (const float*)d_in[11];
    const float* b_lout  = (const float*)d_in[12];
    char*  ws  = (char*)d_ws;
    float* out = (float*)d_out;

    prep_kernel<<<1, 512, 0, stream>>>(numbers, w_num, b_num, W_ih, W_hh,
                                       b_ih, b_hh, W_fh, ws);
    lstm_kernel<<<KOBJ / 16, 256, 0, stream>>>(numbers, b_fh, ws);
    root_kernel<<<1, 384, 0, stream>>>(W_iouh, b_iouh, W_lout, b_lout, ws, out);
}
